// ConditionalMambaNet_29583734735083
// MI455X (gfx1250) — matrix operand order repaired and checked by reading, not compiled, not run on hardware
//
#include <hip/hip_runtime.h>
#include <hip/hip_bf16.h>
#include <math.h>

typedef __attribute__((ext_vector_type(16))) _Float16 v16h;
typedef __attribute__((ext_vector_type(8)))  _Float16 v8h;
typedef __attribute__((ext_vector_type(8)))  float    v8f;
typedef __attribute__((ext_vector_type(4)))  int      v4i;

#if defined(__has_builtin)
#if __has_builtin(__builtin_amdgcn_global_load_async_to_lds_b128) && \
    __has_builtin(__builtin_amdgcn_s_wait_asynccnt)
#define USE_ASYNC_LDS 1
#endif
#endif
#ifndef USE_ASYNC_LDS
#define USE_ASYNC_LDS 0
#endif

namespace {
constexpr int BB   = 2;
constexpr int NTOK = 6400;     // H*W
constexpr int CIN  = 96;
constexpr int HID  = 192;
constexpr int NST  = 16;       // state size N
constexpr int RR   = 12;
constexpr int TT   = 64;       // num classes
constexpr int IRD  = 128;
constexpr int HH   = 80;
constexpr int WWI  = 80;
constexpr int XD   = RR + 2 * NST;   // 44
constexpr int NC   = 32;             // scan chunks per batch
constexpr int CLEN = NTOK / NC;      // 200
constexpr int MTOT = BB * NTOK;      // 12800
}

// ---------------------------------------------------------------------------
// f32 -> f16 bulk conversion (8 elems/thread, b128 in / b128 out)
// ---------------------------------------------------------------------------
__global__ void tohalf_kernel(const float* __restrict__ src,
                              _Float16* __restrict__ dst, int n8) {
  int i = blockIdx.x * blockDim.x + threadIdx.x;
  if (i >= n8) return;
  const float* s = src + i * 8;
  float4 u0 = *(const float4*)(s);
  float4 u1 = *(const float4*)(s + 4);
  v8h h;
  h[0] = (_Float16)u0.x; h[1] = (_Float16)u0.y;
  h[2] = (_Float16)u0.z; h[3] = (_Float16)u0.w;
  h[4] = (_Float16)u1.x; h[5] = (_Float16)u1.y;
  h[6] = (_Float16)u1.z; h[7] = (_Float16)u1.w;
  *(v8h*)(dst + i * 8) = h;
}

// ---------------------------------------------------------------------------
// full_emb[t][j] = sum_i embB[t][i] * tokenW[i][j]   (64 x 16)
// ---------------------------------------------------------------------------
__global__ void fullemb_kernel(const float* __restrict__ embB,
                               const float* __restrict__ tokenW,
                               float* __restrict__ full_emb) {
  int idx = blockIdx.x * blockDim.x + threadIdx.x;
  if (idx >= TT * NST) return;
  int t = idx >> 4, j = idx & 15;
  float s = 0.f;
  for (int i = 0; i < IRD; ++i) s += embB[t * IRD + i] * tokenW[i * NST + j];
  full_emb[idx] = s;
}

// ---------------------------------------------------------------------------
// Routing: argmax over (raw logits + gumbel); det_idx + prompt gather.
// ---------------------------------------------------------------------------
__global__ void route_kernel(const float* __restrict__ x,
                             const float* __restrict__ rw1, const float* __restrict__ rb1,
                             const float* __restrict__ rw2, const float* __restrict__ rb2,
                             const float* __restrict__ gum,
                             const float* __restrict__ full_emb,
                             int* __restrict__ det_idx,
                             float* __restrict__ prompt) {
  __shared__ float hbuf[8][32];
  int w    = threadIdx.x >> 5;
  int lane = threadIdx.x & 31;
  int token = blockIdx.x * 8 + w;                  // grid sized exactly
  const float* xp = x + token * CIN;

  float hs = rb1[lane];
  for (int c = 0; c < CIN; ++c) hs += xp[c] * rw1[lane * CIN + c];
  float h3 = hs * hs * hs;
  float hv = 0.5f * hs * (1.0f + tanhf(0.7978845608028654f * (hs + 0.044715f * h3)));
  hbuf[w][lane] = hv;
  __syncthreads();

  float best = -1e30f; int bidx = 0;
  for (int t2 = 0; t2 < 2; ++t2) {
    int t = lane + 32 * t2;
    float s = rb2[t];
    for (int j = 0; j < 32; ++j) s += hbuf[w][j] * rw2[t * 32 + j];
    float u = gum[token * TT + t];
    float g = -logf(-logf(u));
    float v = s + g;
    if (v > best) { best = v; bidx = t; }
  }
  for (int off = 16; off; off >>= 1) {
    float ov = __shfl_xor(best, off);
    int   oi = __shfl_xor(bidx, off);
    if (ov > best || (ov == best && oi < bidx)) { best = ov; bidx = oi; }
  }
  if (lane == 0) det_idx[token] = bidx;
  if (lane < NST) prompt[token * NST + lane] = full_emb[bidx * NST + lane];
}

// ---------------------------------------------------------------------------
// Stable counting sort of det_idx (keys in [0,64)). One block per batch.
// ---------------------------------------------------------------------------
__global__ void sort_kernel(const int* __restrict__ det_idx,
                            int* __restrict__ sort_idx,
                            int* __restrict__ inv_idx) {
  __shared__ int hist[TT];
  __shared__ int offs[TT];
  int b = blockIdx.x;
  const int* di = det_idx + b * NTOK;
  if (threadIdx.x < TT) hist[threadIdx.x] = 0;
  __syncthreads();
  for (int i = threadIdx.x; i < NTOK; i += blockDim.x) atomicAdd(&hist[di[i]], 1);
  __syncthreads();
  if (threadIdx.x == 0) {
    int acc = 0;
    for (int k = 0; k < TT; ++k) { offs[k] = acc; acc += hist[k]; }
  }
  __syncthreads();
  if (threadIdx.x < TT) {
    int k = threadIdx.x;
    int pos = offs[k];
    for (int i = 0; i < NTOK; ++i) {
      if (di[i] == k) {
        sort_idx[b * NTOK + pos] = i;
        inv_idx[b * NTOK + i]    = pos;
        ++pos;
      }
    }
  }
}

// ---------------------------------------------------------------------------
// WMMA GEMM:  C[M x N] = A[M x K] * W^T + bias; A,W pre-converted f16.
// BM=64, BN=64, BK=32; 256 threads (8 waves). LDS tiles pre-permuted in WMMA
// fragment order: each lane reads its whole v16h fragment as one contiguous
// 32-byte LDS access; each thread stages one aligned 16-byte chunk of A and
// one of B per K-step via async global->LDS DMA (ASYNCcnt) when available.
//   A frag (lane group hi): k(e) = ((e<8)? e : e+8) + 8*hi
//     chunk q (e=8q+j): global k = k0 + 16*q + 8*hi + j   (8 consecutive)
//   B frag: k(e) = ((e<8)? e : e+8) + 8*hi
//     chunk q:          global k = k0 + 16*q + 8*hi + j   (8 consecutive)
// M multiple of 64, K multiple of 32; N guarded/zero-padded.
// ---------------------------------------------------------------------------
__global__ __launch_bounds__(256) void gemm_wmma_f16(
    const _Float16* __restrict__ A, const _Float16* __restrict__ W,
    const float* __restrict__ bias, float* __restrict__ Cout,
    int M, int N, int K) {
  __shared__ v16h As2[64][2];   // [row][hi]  4KB
  __shared__ v16h Bs2[64][2];   // [col][hi]  4KB
  v8h* As8 = (v8h*)As2;
  v8h* Bs8 = (v8h*)Bs2;

  int tid  = threadIdx.x;
  int wave = tid >> 5;
  int lane = tid & 31;
  int m0 = blockIdx.x * 64;
  int n0 = blockIdx.y * 64;
  int wm = wave & 3;     // m-tile (16 rows)
  int wn = wave >> 2;    // n-half (32 cols)
  int col = lane & 15, hi = lane >> 4;

  // staging decomposition: each thread owns one 8-half chunk of A and of B
  int sr = tid >> 2;          // 0..63 : row (A) / col (B)
  int srem = tid & 3;
  int sh = srem >> 1;         // hi group
  int sq = srem & 1;          // chunk within fragment

  const int aoff = (m0 + sr) * K + 16 * sq + 8 * sh;   // + k0
  const bool bvalid = (n0 + sr) < N;
  const int boff = bvalid ? ((n0 + sr) * K + 16 * sq + 8 * sh) : 0;
  const int sdst = (sr * 2 + sh) * 2 + sq;

  // zero-fill B slots for padded columns once; never overwritten below
  if (!bvalid) Bs8[sdst] = (v8h){};

  v8f acc0 = {}; v8f acc1 = {};

  for (int k0 = 0; k0 < K; k0 += 32) {
#if USE_ASYNC_LDS
    __builtin_amdgcn_global_load_async_to_lds_b128(
        (v4i*)(A + aoff + k0), (v4i*)&As8[sdst], 0, 0);
    if (bvalid)
      __builtin_amdgcn_global_load_async_to_lds_b128(
          (v4i*)(W + boff + k0), (v4i*)&Bs8[sdst], 0, 0);
    __builtin_amdgcn_s_wait_asynccnt(0);
#else
    As8[sdst] = *(const v8h*)(A + aoff + k0);
    if (bvalid) Bs8[sdst] = *(const v8h*)(W + boff + k0);
#endif
    if (k0 + 32 < K) __builtin_prefetch(A + aoff + k0 + 32, 0, 1);
    __syncthreads();

    v16h afrag = As2[wm * 16 + col][hi];
    v16h b0    = Bs2[wn * 32 + col][hi];
    v16h b1    = Bs2[wn * 32 + 16 + col][hi];
    acc0 = __builtin_amdgcn_wmma_f32_16x16x32_f16(false, afrag, false, b0,
                                                  (short)0, acc0, false, false);
    acc1 = __builtin_amdgcn_wmma_f32_16x16x32_f16(false, afrag, false, b1,
                                                  (short)0, acc1, false, false);
    __syncthreads();
  }

  int ccol0 = n0 + wn * 32 + col;
  int ccol1 = ccol0 + 16;
  float bv0 = (ccol0 < N && bias) ? bias[ccol0] : 0.f;
  float bv1 = (ccol1 < N && bias) ? bias[ccol1] : 0.f;
#pragma unroll
  for (int j = 0; j < 8; ++j) {
    int r = m0 + wm * 16 + 8 * hi + j;
    if (ccol0 < N) Cout[r * N + ccol0] = acc0[j] + bv0;
    if (ccol1 < N) Cout[r * N + ccol1] = acc1[j] + bv1;
  }
}

// ---------------------------------------------------------------------------
// Depthwise 3x3 CPE conv + sigmoid gate: fg = f * sigmoid(conv(f) + b)
// ---------------------------------------------------------------------------
__global__ void cpe_kernel(const float* __restrict__ f,
                           const float* __restrict__ cw,
                           const float* __restrict__ cb,
                           float* __restrict__ fg) {
  int idx = blockIdx.x * blockDim.x + threadIdx.x;
  if (idx >= MTOT * HID) return;
  int c = idx % HID;
  int p = idx / HID;
  int xc = p % WWI;
  int p2 = p / WWI;
  int yr = p2 % HH;
  int b  = p2 / HH;
  float s = cb[c];
#pragma unroll
  for (int ky = 0; ky < 3; ++ky)
#pragma unroll
    for (int kx = 0; kx < 3; ++kx) {
      int yy = yr + ky - 1, xx = xc + kx - 1;
      if (yy >= 0 && yy < HH && xx >= 0 && xx < WWI)
        s += cw[c * 9 + ky * 3 + kx] * f[(b * NTOK + yy * WWI + xx) * HID + c];
    }
  float v = f[idx];
  fg[idx] = v * (1.0f / (1.0f + __expf(-s)));
}

// ---------------------------------------------------------------------------
// xs[b, l, :] = fg[b, sort_idx[b,l], :]
// ---------------------------------------------------------------------------
__global__ void gather_kernel(const float* __restrict__ fg,
                              const int* __restrict__ sort_idx,
                              float* __restrict__ xs) {
  int idx = blockIdx.x * blockDim.x + threadIdx.x;
  if (idx >= MTOT * HID) return;
  int d = idx % HID;
  int t = idx / HID;
  int l = t % NTOK;
  int b = t / NTOK;
  int src = sort_idx[b * NTOK + l];
  xs[idx] = fg[(b * NTOK + src) * HID + d];
}

// ---------------------------------------------------------------------------
// Scan phase A: per-chunk local recurrence carries (P = prod dA, S = local end)
// ---------------------------------------------------------------------------
__global__ __launch_bounds__(192) void scanA_kernel(
    const float* __restrict__ xs, const float* __restrict__ xdbl,
    const float* __restrict__ dt_w, const float* __restrict__ dt_b,
    const float* __restrict__ A_logs,
    float* __restrict__ carryP, float* __restrict__ carryS) {
  __shared__ float srow[16][RR + NST];   // dts(12) + Bs(16)
  int b  = blockIdx.x / NC;
  int ch = blockIdx.x % NC;
  int d  = threadIdx.x;
  int base = b * NTOK + ch * CLEN;

  float negA[NST];
#pragma unroll
  for (int nn = 0; nn < NST; ++nn) negA[nn] = -__expf(A_logs[d * NST + nn]);
  float dtw[RR];
#pragma unroll
  for (int r = 0; r < RR; ++r) dtw[r] = dt_w[d * RR + r];
  float dtb = dt_b[d];

  float P[NST], S[NST];
#pragma unroll
  for (int nn = 0; nn < NST; ++nn) { P[nn] = 1.f; S[nn] = 0.f; }

  for (int s0 = 0; s0 < CLEN; s0 += 16) {
    int nsteps = (CLEN - s0) < 16 ? (CLEN - s0) : 16;
    for (int i = d; i < nsteps * (RR + NST); i += HID) {
      int st = i / (RR + NST), c = i % (RR + NST);
      srow[st][c] = xdbl[(base + s0 + st) * XD + c];
    }
    __syncthreads();
    for (int st = 0; st < nsteps; ++st) {
      int l = base + s0 + st;
      float xv = xs[l * HID + d];
      float draw = dtb;
#pragma unroll
      for (int r = 0; r < RR; ++r) draw += srow[st][r] * dtw[r];
      float delta = (draw > 20.f) ? draw : log1pf(__expf(draw));
#pragma unroll
      for (int nn = 0; nn < NST; ++nn) {
        float dA  = __expf(delta * negA[nn]);
        float dBu = delta * srow[st][RR + nn] * xv;
        P[nn] *= dA;
        S[nn] = dA * S[nn] + dBu;
      }
    }
    __syncthreads();
  }
  int co = ((b * NC + ch) * HID + d) * NST;
#pragma unroll
  for (int nn = 0; nn < NST; ++nn) { carryP[co + nn] = P[nn]; carryS[co + nn] = S[nn]; }
}

// ---------------------------------------------------------------------------
// Scan phase B: sequentially combine chunk carries -> carry-in per chunk.
// ---------------------------------------------------------------------------
__global__ void scanB_kernel(const float* __restrict__ carryP,
                             const float* __restrict__ carryS,
                             float* __restrict__ carryIn) {
  int idx = blockIdx.x * blockDim.x + threadIdx.x;   // BB*HID*NST = 6144
  if (idx >= BB * HID * NST) return;
  int b  = idx / (HID * NST);
  int dn = idx % (HID * NST);
  float st = 0.f;
  for (int ch = 0; ch < NC; ++ch) {
    int o = (b * NC + ch) * (HID * NST) + dn;
    carryIn[o] = st;
    st = carryP[o] * st + carryS[o];
  }
}

// ---------------------------------------------------------------------------
// Scan phase C: replay local scan with carry-in; y = states . Cs + xs*Ds.
// Cs = x_dbl[...,28:44] + prompt (prompt stays in ORIGINAL order, per ref).
// ---------------------------------------------------------------------------
__global__ __launch_bounds__(192) void scanC_kernel(
    const float* __restrict__ xs, const float* __restrict__ xdbl,
    const float* __restrict__ prompt,
    const float* __restrict__ dt_w, const float* __restrict__ dt_b,
    const float* __restrict__ A_logs, const float* __restrict__ Ds,
    const float* __restrict__ carryIn, float* __restrict__ y) {
  __shared__ float srow[16][XD];   // dts(12) + Bs(16) + Cs(16)
  int b  = blockIdx.x / NC;
  int ch = blockIdx.x % NC;
  int d  = threadIdx.x;
  int base = b * NTOK + ch * CLEN;

  float negA[NST];
#pragma unroll
  for (int nn = 0; nn < NST; ++nn) negA[nn] = -__expf(A_logs[d * NST + nn]);
  float dtw[RR];
#pragma unroll
  for (int r = 0; r < RR; ++r) dtw[r] = dt_w[d * RR + r];
  float dtb = dt_b[d];
  float dsv = Ds[d];

  float S[NST];
  int co = ((b * NC + ch) * HID + d) * NST;
#pragma unroll
  for (int nn = 0; nn < NST; ++nn) S[nn] = carryIn[co + nn];

  for (int s0 = 0; s0 < CLEN; s0 += 16) {
    int nsteps = (CLEN - s0) < 16 ? (CLEN - s0) : 16;
    for (int i = d; i < nsteps * XD; i += HID) {
      int st = i / XD, c = i % XD;
      int l = base + s0 + st;
      float v = xdbl[l * XD + c];
      if (c >= RR + NST) v += prompt[l * NST + (c - RR - NST)];
      srow[st][c] = v;
    }
    __syncthreads();
    for (int st = 0; st < nsteps; ++st) {
      int l = base + s0 + st;
      float xv = xs[l * HID + d];
      float draw = dtb;
#pragma unroll
      for (int r = 0; r < RR; ++r) draw += srow[st][r] * dtw[r];
      float delta = (draw > 20.f) ? draw : log1pf(__expf(draw));
      float acc = 0.f;
#pragma unroll
      for (int nn = 0; nn < NST; ++nn) {
        float dA  = __expf(delta * negA[nn]);
        float dBu = delta * srow[st][RR + nn] * xv;
        S[nn] = dA * S[nn] + dBu;
        acc += S[nn] * srow[st][RR + NST + nn];
      }
      y[l * HID + d] = acc + xv * dsv;
    }
    __syncthreads();
  }
}

// ---------------------------------------------------------------------------
// LayerNorm over hid=192. One wave per token, 6 elems/lane.
// ---------------------------------------------------------------------------
__global__ void ln_kernel(const float* __restrict__ y,
                          const float* __restrict__ g,
                          const float* __restrict__ bta,
                          float* __restrict__ yn) {
  int token = blockIdx.x * 8 + (threadIdx.x >> 5);
  int lane  = threadIdx.x & 31;
  const float* yp = y + token * HID;
  float v[6], s = 0.f, s2 = 0.f;
#pragma unroll
  for (int i = 0; i < 6; ++i) {
    v[i] = yp[lane + 32 * i];
    s += v[i]; s2 += v[i] * v[i];
  }
  for (int off = 16; off; off >>= 1) {
    s  += __shfl_xor(s, off);
    s2 += __shfl_xor(s2, off);
  }
  float mu  = s * (1.0f / HID);
  float var = s2 * (1.0f / HID) - mu * mu;
  float inv = rsqrtf(var + 1e-5f);
#pragma unroll
  for (int i = 0; i < 6; ++i) {
    int dd = lane + 32 * i;
    yn[token * HID + dd] = (v[i] - mu) * inv * g[dd] + bta[dd];
  }
}

// ---------------------------------------------------------------------------
// Un-sort: out[b, i, :] = out_sorted[b, inv_idx[b,i], :]
// ---------------------------------------------------------------------------
__global__ void scatter_kernel(const float* __restrict__ outs,
                               const int* __restrict__ inv_idx,
                               float* __restrict__ out) {
  int idx = blockIdx.x * blockDim.x + threadIdx.x;
  if (idx >= MTOT * CIN) return;
  int c = idx % CIN;
  int t = idx / CIN;
  int i = t % NTOK;
  int b = t / NTOK;
  int pos = inv_idx[b * NTOK + i];
  out[idx] = outs[(b * NTOK + pos) * CIN + c];
}

// ---------------------------------------------------------------------------
extern "C" void kernel_launch(void* const* d_in, const int* in_sizes, int n_in,
                              void* d_out, int out_size, void* d_ws, size_t ws_size,
                              hipStream_t stream) {
  const float* x        = (const float*)d_in[0];
  const float* tokenW   = (const float*)d_in[1];
  const float* embB     = (const float*)d_in[2];
  const float* route_w1 = (const float*)d_in[3];
  const float* route_b1 = (const float*)d_in[4];
  const float* route_w2 = (const float*)d_in[5];
  const float* route_b2 = (const float*)d_in[6];
  const float* inproj_w = (const float*)d_in[7];
  const float* inproj_b = (const float*)d_in[8];
  const float* cpe_w    = (const float*)d_in[9];
  const float* cpe_b    = (const float*)d_in[10];
  const float* xproj_w  = (const float*)d_in[11];
  const float* dt_w     = (const float*)d_in[12];
  const float* dt_b     = (const float*)d_in[13];
  const float* A_logs   = (const float*)d_in[14];
  const float* Ds       = (const float*)d_in[15];
  const float* ln_g     = (const float*)d_in[16];
  const float* ln_b     = (const float*)d_in[17];
  const float* outprj_w = (const float*)d_in[18];
  const float* outprj_b = (const float*)d_in[19];
  const float* gumbel_u = (const float*)d_in[20];
  float* out = (float*)d_out;

  // ---- workspace bump allocation ----
  char* base = (char*)d_ws;
  size_t off = 0;
  auto alloc_f = [&](size_t nelem) -> float* {
    float* p = (float*)(base + off);
    off += ((nelem * sizeof(float) + 255) / 256) * 256;
    return p;
  };
  auto alloc_h = [&](size_t nelem) -> _Float16* {
    _Float16* p = (_Float16*)(base + off);
    off += ((nelem * sizeof(_Float16) + 255) / 256) * 256;
    return p;
  };
  float* full_emb = alloc_f(TT * NST);
  float* prompt   = alloc_f((size_t)MTOT * NST);
  float* bufA     = alloc_f((size_t)MTOT * HID);  // f -> xs -> yn
  float* bufB     = alloc_f((size_t)MTOT * HID);  // fg -> y -> out_sorted
  float* xdbl     = alloc_f((size_t)MTOT * XD);
  float* carryP   = alloc_f((size_t)BB * NC * HID * NST);
  float* carryS   = alloc_f((size_t)BB * NC * HID * NST);
  float* carryIn  = alloc_f((size_t)BB * NC * HID * NST);
  _Float16* xh  = alloc_h((size_t)MTOT * CIN);   // x in f16
  _Float16* ah  = alloc_h((size_t)MTOT * HID);   // xs / yn in f16
  _Float16* wih = alloc_h((size_t)HID * CIN);    // inproj_w f16
  _Float16* wxh = alloc_h((size_t)XD * HID);     // xproj_w f16
  _Float16* woh = alloc_h((size_t)CIN * HID);    // outproj_w f16
  int* det_idx  = (int*)(base + off); off += ((size_t)MTOT * 4 + 255) / 256 * 256;
  int* sort_idx = (int*)(base + off); off += ((size_t)MTOT * 4 + 255) / 256 * 256;
  int* inv_idx  = (int*)(base + off); off += ((size_t)MTOT * 4 + 255) / 256 * 256;

  // 0) one-time f16 conversions of GEMM operands
  tohalf_kernel<<<(MTOT * CIN / 8 + 255) / 256, 256, 0, stream>>>(x, xh, MTOT * CIN / 8);
  tohalf_kernel<<<(HID * CIN / 8 + 255) / 256, 256, 0, stream>>>(inproj_w, wih, HID * CIN / 8);
  tohalf_kernel<<<(XD * HID / 8 + 255) / 256, 256, 0, stream>>>(xproj_w, wxh, XD * HID / 8);
  tohalf_kernel<<<(CIN * HID / 8 + 255) / 256, 256, 0, stream>>>(outprj_w, woh, CIN * HID / 8);

  // 1) full_emb = embB @ tokenW
  fullemb_kernel<<<4, 256, 0, stream>>>(embB, tokenW, full_emb);

  // 2) routing + gumbel argmax + prompt gather
  route_kernel<<<MTOT / 8, 256, 0, stream>>>(x, route_w1, route_b1, route_w2,
                                             route_b2, gumbel_u, full_emb,
                                             det_idx, prompt);

  // 3) stable counting sort per batch
  sort_kernel<<<BB, 256, 0, stream>>>(det_idx, sort_idx, inv_idx);

  // 4) inproj GEMM: f = x @ inproj_w^T + b   (12800x96 * 96x192)
  gemm_wmma_f16<<<dim3(MTOT / 64, HID / 64), 256, 0, stream>>>(
      xh, wih, inproj_b, bufA, MTOT, HID, CIN);

  // 5) CPE depthwise conv + sigmoid gate
  cpe_kernel<<<(MTOT * HID + 255) / 256, 256, 0, stream>>>(bufA, cpe_w, cpe_b, bufB);

  // 6) gather into sorted order: xs (overwrites bufA)
  gather_kernel<<<(MTOT * HID + 255) / 256, 256, 0, stream>>>(bufB, sort_idx, bufA);

  // 7) xproj GEMM: x_dbl = xs @ xproj_w^T    (12800x192 * 192x44)
  tohalf_kernel<<<(MTOT * HID / 8 + 255) / 256, 256, 0, stream>>>(bufA, ah, MTOT * HID / 8);
  gemm_wmma_f16<<<dim3(MTOT / 64, 1), 256, 0, stream>>>(
      ah, wxh, nullptr, xdbl, MTOT, XD, HID);

  // 8) chunked selective scan (fused delta/softplus/exp), y into bufB
  scanA_kernel<<<BB * NC, HID, 0, stream>>>(bufA, xdbl, dt_w, dt_b, A_logs,
                                            carryP, carryS);
  scanB_kernel<<<(BB * HID * NST + 255) / 256, 256, 0, stream>>>(carryP, carryS, carryIn);
  scanC_kernel<<<BB * NC, HID, 0, stream>>>(bufA, xdbl, prompt, dt_w, dt_b,
                                            A_logs, Ds, carryIn, bufB);

  // 9) layernorm: yn into bufA
  ln_kernel<<<MTOT / 8, 256, 0, stream>>>(bufB, ln_g, ln_b, bufA);

  // 10) outproj GEMM: out_sorted = yn @ outproj_w^T + b  (into bufB)
  tohalf_kernel<<<(MTOT * HID / 8 + 255) / 256, 256, 0, stream>>>(bufA, ah, MTOT * HID / 8);
  gemm_wmma_f16<<<dim3(MTOT / 64, (CIN + 63) / 64), 256, 0, stream>>>(
      ah, woh, outprj_b, bufB, MTOT, CIN, HID);

  // 11) un-sort to original token order
  scatter_kernel<<<(MTOT * CIN + 255) / 256, 256, 0, stream>>>(bufB, inv_idx, out);
}
